// NetworkModel_34428457844740
// MI455X (gfx1250) — matrix operand order repaired and checked by reading, not compiled, not run on hardware
//
#include <hip/hip_runtime.h>
#include <hip/hip_bf16.h>
#include <math.h>

typedef _Float16 v16h __attribute__((ext_vector_type(16)));
typedef _Float16 v8h  __attribute__((ext_vector_type(8)));
typedef float    v8f  __attribute__((ext_vector_type(8)));

#define HEADS 10
#define NEG_SLOPE 0.2f

// ---------------------------------------------------------------------------
// WMMA GEMM: C[M,Nc] = act(A[M,K] @ B[K,Nc] + bias)
// f32 in -> f16 LDS tiles -> v_wmma_f32_16x16x32_f16 (f32 accumulate).
// Block = 128 threads = 4 waves. Block tile: 16 rows x 64 cols (16 cols/wave).
// K is zero-padded to KMAX (multiple of 32) in LDS. Requires M % 16 == 0.
// act: 0 = identity, 1 = ELU.
// ---------------------------------------------------------------------------
template<int KMAX>
__global__ __launch_bounds__(128)
void gemm_wmma_f16(const float* __restrict__ A, const float* __restrict__ B,
                   const float* __restrict__ bias, float* __restrict__ C,
                   int M, int K, int Nc, int act)
{
    __shared__ __align__(64) _Float16 Ah[16 * KMAX];   // A tile, row-major [16][KMAX]
    __shared__ __align__(64) _Float16 Bt[64 * KMAX];   // B tile transposed [64][KMAX]

    const int row0 = blockIdx.x * 16;
    const int col0 = blockIdx.y * 64;
    const int t    = threadIdx.x;

    for (int idx = t; idx < 16 * KMAX; idx += 128) {
        int r = idx / KMAX, k = idx % KMAX;
        Ah[idx] = (k < K) ? (_Float16)A[(size_t)(row0 + r) * K + k] : (_Float16)0.f;
    }
    for (int idx = t; idx < 64 * KMAX; idx += 128) {
        int n = idx / KMAX, k = idx % KMAX;
        int col = col0 + n;
        Bt[idx] = (k < K && col < Nc) ? (_Float16)B[(size_t)k * Nc + col] : (_Float16)0.f;
    }
    __syncthreads();

    const int wave = t >> 5;
    const int lane = t & 31;
    const int hi   = (lane >= 16) ? 1 : 0;   // high half-wave
    const int mr   = lane & 15;              // M index (A) / N index (B,C)

    const _Float16* arow = &Ah[mr * KMAX];
    const _Float16* brow = &Bt[(wave * 16 + mr) * KMAX];

    v8f acc = {};
    #pragma unroll
    for (int k0 = 0; k0 < KMAX; k0 += 32) {
        // 16-bit A 16x32 layout: lanes 0-15 hold K {0..7,16..23}, lanes 16-31 hold K {8..15,24..31}
        union { v16h v; v8h h[2]; } av;
        const _Float16* ap = arow + k0 + (hi ? 8 : 0);
        av.h[0] = *(const v8h*)(ap);
        av.h[1] = *(const v8h*)(ap + 16);
        // 16-bit B 32x16 layout: lanes 0-15 hold K {0..7,16..23}, lanes 16-31 hold K {8..15,24..31} (col = lane&15)
        union { v16h v; v8h h[2]; } bu;
        const _Float16* bp = brow + k0 + (hi ? 8 : 0);
        bu.h[0] = *(const v8h*)(bp);
        bu.h[1] = *(const v8h*)(bp + 16);
        v16h bv = bu.v;
        acc = __builtin_amdgcn_wmma_f32_16x16x32_f16(false, av.v, false, bv,
                                                     (short)0, acc, false, false);
    }

    // C/D layout: VGPR r -> row r (lanes 0-15) or r+8 (lanes 16-31), col = lane&15
    const int col = col0 + wave * 16 + mr;
    if (col < Nc) {
        const float bb = bias ? bias[col] : 0.f;
        #pragma unroll
        for (int r = 0; r < 8; ++r) {
            int row = row0 + r + (hi ? 8 : 0);
            float v = acc[r] + bb;
            if (act == 1) v = v > 0.f ? v : expf(v) - 1.f;   // ELU (alpha=1)
            C[(size_t)row * Nc + col] = v;
        }
    }
}

// ---------------------------------------------------------------------------
// CSR build (dst-indexed), done once per launch; graph is static across layers
// ---------------------------------------------------------------------------
__global__ void init_int_kernel(int* __restrict__ p, int v, int n)
{
    int i = blockIdx.x * blockDim.x + threadIdx.x;
    if (i < n) p[i] = v;
}

__global__ void count_kernel(const int* __restrict__ dst, int* __restrict__ deg,
                             int E, int Etot)
{
    int i = blockIdx.x * blockDim.x + threadIdx.x;
    if (i >= Etot) return;
    int d = (i < E) ? dst[i] : (i - E);   // self-loops appended
    atomicAdd(&deg[d], 1);
}

// single-block exclusive scan: rowptr[0..N], rowptr[N] = total
__global__ void scan_kernel(const int* __restrict__ deg, int* __restrict__ rowptr, int N)
{
    __shared__ int buf[256];
    int carry = 0;
    for (int base = 0; base < N; base += 256) {
        int i = base + (int)threadIdx.x;
        int v = (i < N) ? deg[i] : 0;
        buf[threadIdx.x] = v;
        __syncthreads();
        #pragma unroll
        for (int ofs = 1; ofs < 256; ofs <<= 1) {
            int tv = (threadIdx.x >= (unsigned)ofs) ? buf[threadIdx.x - ofs] : 0;
            __syncthreads();
            buf[threadIdx.x] += tv;
            __syncthreads();
        }
        if (i < N) rowptr[i] = carry + buf[threadIdx.x] - v;   // exclusive
        carry += buf[255];
        __syncthreads();
    }
    if (threadIdx.x == 0) rowptr[N] = carry;
}

__global__ void copy_int_kernel(const int* __restrict__ a, int* __restrict__ b, int n)
{
    int i = blockIdx.x * blockDim.x + threadIdx.x;
    if (i < n) b[i] = a[i];
}

__global__ void bucket_kernel(const int* __restrict__ dst, int* __restrict__ wpos,
                              int* __restrict__ eidx, int E, int Etot)
{
    int i = blockIdx.x * blockDim.x + threadIdx.x;
    if (i >= Etot) return;
    int d = (i < E) ? dst[i] : (i - E);
    int p = atomicAdd(&wpos[d], 1);
    eidx[p] = i;
}

// ---------------------------------------------------------------------------
// GAT attention (gather formulation, no atomics)
// ---------------------------------------------------------------------------
__device__ __forceinline__ float leaky(float x) { return x > 0.f ? x : NEG_SLOPE * x; }

// e_s[n,h] = sum_c hh[n,h,c]*a_src[h,c] ; e_d likewise
__global__ void attn_scores_kernel(const float* __restrict__ hh,
                                   const float* __restrict__ a_src,
                                   const float* __restrict__ a_dst,
                                   float* __restrict__ es, float* __restrict__ ed,
                                   int N, int C)
{
    int i = blockIdx.x * blockDim.x + threadIdx.x;   // n*HEADS + h
    if (i >= N * HEADS) return;
    int h = i % HEADS, n = i / HEADS;
    const float* hp = hh + ((size_t)n * HEADS + h) * C;
    const float* as = a_src + h * C;
    const float* ad = a_dst + h * C;
    float s = 0.f, d = 0.f;
    for (int c = 0; c < C; ++c) { float v = hp[c]; s += v * as[c]; d += v * ad[c]; }
    es[i] = s; ed[i] = d;
}

// Per (dst,head): 3-pass softmax over incident edges -> normalized alpha[j,h]
__global__ void node_softmax_kernel(const int* __restrict__ rowptr, const int* __restrict__ eidx,
                                    const int* __restrict__ src,
                                    const float* __restrict__ es, const float* __restrict__ ed,
                                    float* __restrict__ alpha, int E, int N)
{
    int t = blockIdx.x * blockDim.x + threadIdx.x;   // d*HEADS + h
    if (t >= N * HEADS) return;
    int h = t % HEADS, d = t / HEADS;
    int beg = rowptr[d], end = rowptr[d + 1];
    float edh = ed[d * HEADS + h];

    float mx = -INFINITY;
    for (int j = beg; j < end; ++j) {
        int i = eidx[j];
        int s = (i < E) ? src[i] : (i - E);
        mx = fmaxf(mx, leaky(es[s * HEADS + h] + edh));
    }
    // every node has a self-loop => mx is finite (matches reference's isfinite fix)
    float sum = 0.f;
    for (int j = beg; j < end; ++j) {
        int i = eidx[j];
        int s = (i < E) ? src[i] : (i - E);
        float p = expf(leaky(es[s * HEADS + h] + edh) - mx);
        sum += p;
        alpha[(size_t)j * HEADS + h] = p;
    }
    float inv = 1.f / (sum + 1e-16f);
    for (int j = beg; j < end; ++j) alpha[(size_t)j * HEADS + h] *= inv;
}

// y[d,c] = elu( (1/H) * sum_h sum_{j in in(d)} alpha[j,h]*hh[src_j,h,c] + bias[c] )
__global__ void gat_aggregate_kernel(const int* __restrict__ rowptr, const int* __restrict__ eidx,
                                     const int* __restrict__ src,
                                     const float* __restrict__ alpha, const float* __restrict__ hh,
                                     const float* __restrict__ bias, float* __restrict__ y,
                                     int E, int N, int C)
{
    int t = blockIdx.x * blockDim.x + threadIdx.x;   // d*C + c
    if (t >= N * C) return;
    int c = t % C, d = t / C;
    int beg = rowptr[d], end = rowptr[d + 1];
    float acc = 0.f;
    for (int j = beg; j < end; ++j) {
        int i = eidx[j];
        int s = (i < E) ? src[i] : (i - E);
        const float* hs = hh + (size_t)s * HEADS * C + c;
        const float* al = alpha + (size_t)j * HEADS;
        if (j + 1 < end) {   // prefetch next edge's feature row (global_prefetch_b8)
            int i2 = eidx[j + 1];
            int s2 = (i2 < E) ? src[i2] : (i2 - E);
            __builtin_prefetch(hh + (size_t)s2 * HEADS * C + c, 0, 3);
        }
        #pragma unroll
        for (int h = 0; h < HEADS; ++h) acc += al[h] * hs[h * C];
    }
    float v = acc * (1.f / HEADS) + bias[c];
    y[t] = v > 0.f ? v : expf(v) - 1.f;   // fused ELU
}

__global__ void log_softmax_kernel(const float* __restrict__ logits, float* __restrict__ out,
                                   int N, int CLS)
{
    int n = blockIdx.x * blockDim.x + threadIdx.x;
    if (n >= N) return;
    const float* r = logits + (size_t)n * CLS;
    float mx = -INFINITY;
    for (int c = 0; c < CLS; ++c) mx = fmaxf(mx, r[c]);
    float s = 0.f;
    for (int c = 0; c < CLS; ++c) s += expf(r[c] - mx);
    float lse = mx + logf(s);
    for (int c = 0; c < CLS; ++c) out[(size_t)n * CLS + c] = r[c] - lse;
}

// ---------------------------------------------------------------------------
// Launcher
// ---------------------------------------------------------------------------
extern "C" void kernel_launch(void* const* d_in, const int* in_sizes, int n_in,
                              void* d_out, int out_size, void* d_ws, size_t ws_size,
                              hipStream_t stream)
{
    const float* x   = (const float*)d_in[0];
    const int*   ei  = (const int*)d_in[1];     // [2,E] edge_index (int32 per harness)
    const float* W0  = (const float*)d_in[3];
    const float* b0  = (const float*)d_in[4];
    const float* Wc1 = (const float*)d_in[5];
    const float* as1 = (const float*)d_in[6];
    const float* ad1 = (const float*)d_in[7];
    const float* bc1 = (const float*)d_in[8];
    const float* Wc2 = (const float*)d_in[9];
    const float* as2 = (const float*)d_in[10];
    const float* ad2 = (const float*)d_in[11];
    const float* bc2 = (const float*)d_in[12];
    const float* Wc3 = (const float*)d_in[13];
    const float* as3 = (const float*)d_in[14];
    const float* ad3 = (const float*)d_in[15];
    const float* bc3 = (const float*)d_in[16];
    const float* W1  = (const float*)d_in[17];
    const float* b1  = (const float*)d_in[18];
    const float* W2  = (const float*)d_in[19];
    const float* b2  = (const float*)d_in[20];

    const int N    = in_sizes[2];          // #nodes (10000)
    const int E    = in_sizes[1] / 2;      // 160000
    const int Etot = E + N;                // + self-loops
    const int F    = in_sizes[0] / N;      // 24
    const int CLS  = in_sizes[20];         // 24

    const int* srcI = ei;
    const int* dstI = ei + E;

    // Workspace carve-up. All buffers fully (re)written each call before use.
    float* ws = (float*)d_ws;
    size_t off = 0;
    auto carveF = [&](size_t nelem) { float* p = ws + off; off += nelem; return p; };
    float* hh    = carveF((size_t)N * HEADS * 128);  // per-head features (max C=128)
    float* y0    = carveF((size_t)N * 256);          // ping
    float* y1    = carveF((size_t)N * 256);          // pong
    float* es    = carveF((size_t)N * HEADS);
    float* ed    = carveF((size_t)N * HEADS);
    float* alpha = carveF((size_t)Etot * HEADS);

    int* ib = (int*)(ws + off);
    size_t ioff = 0;
    auto carveI = [&](size_t nelem) { int* p = ib + ioff; ioff += nelem; return p; };
    int* deg    = carveI(N);
    int* rowptr = carveI(N + 1);
    int* wpos   = carveI(N);
    int* eidx   = carveI(Etot);

    // ---- CSR build (once; reused by all 3 GAT layers) ----
    init_int_kernel<<<(N + 255) / 256, 256, 0, stream>>>(deg, 0, N);
    count_kernel<<<(Etot + 255) / 256, 256, 0, stream>>>(dstI, deg, E, Etot);
    scan_kernel<<<1, 256, 0, stream>>>(deg, rowptr, N);
    copy_int_kernel<<<(N + 255) / 256, 256, 0, stream>>>(rowptr, wpos, N);
    bucket_kernel<<<(Etot + 255) / 256, 256, 0, stream>>>(dstI, wpos, eidx, E, Etot);

    auto gemm = [&](const float* A, const float* B, const float* bias, float* Cp,
                    int M, int K, int Nc, int act) {
        dim3 grid(M / 16, (Nc + 63) / 64);
        if (K <= 32)
            gemm_wmma_f16<32><<<grid, 128, 0, stream>>>(A, B, bias, Cp, M, K, Nc, act);
        else if (K <= 64)
            gemm_wmma_f16<64><<<grid, 128, 0, stream>>>(A, B, bias, Cp, M, K, Nc, act);
        else if (K <= 128)
            gemm_wmma_f16<128><<<grid, 128, 0, stream>>>(A, B, bias, Cp, M, K, Nc, act);
        else
            gemm_wmma_f16<256><<<grid, 128, 0, stream>>>(A, B, bias, Cp, M, K, Nc, act);
    };

    auto gat = [&](const float* yin, int Fin, const float* Wc, const float* asrc,
                   const float* adst, const float* bc, int C, float* yout) {
        gemm(yin, Wc, nullptr, hh, N, Fin, HEADS * C, 0);   // per-head features via WMMA
        const int nh = N * HEADS;
        attn_scores_kernel<<<(nh + 255) / 256, 256, 0, stream>>>(hh, asrc, adst, es, ed, N, C);
        node_softmax_kernel<<<(nh + 255) / 256, 256, 0, stream>>>(rowptr, eidx, srcI, es, ed, alpha, E, N);
        gat_aggregate_kernel<<<(N * C + 255) / 256, 256, 0, stream>>>(rowptr, eidx, srcI, alpha,
                                                                      hh, bc, yout, E, N, C);
    };

    // h = elu(x @ W0 + b0)
    gemm(x, W0, b0, y0, N, F, 16, 1);
    // three GAT layers (ELU fused into aggregation epilogue)
    gat(y0, 16,  Wc1, as1, ad1, bc1, 32,  y1);
    gat(y1, 32,  Wc2, as2, ad2, bc2, 64,  y0);
    gat(y0, 64,  Wc3, as3, ad3, bc3, 128, y1);
    // MLP head
    gemm(y1, W1, b1, y0, N, 128, 256, 1);        // elu
    gemm(y0, W2, b2, hh, N, 256, CLS, 0);        // logits (reuse hh)
    log_softmax_kernel<<<(N + 255) / 256, 256, 0, stream>>>(hh, (float*)d_out, N, CLS);
}
